// ModuleSepconv_50826642981221
// MI455X (gfx1250) — matrix operand order repaired and checked by reading, not compiled, not run on hardware
//
#include <hip/hip_runtime.h>
#include <hip/hip_bf16.h>

typedef __bf16 bf16_t;
typedef __attribute__((ext_vector_type(16))) __bf16 v16bf;
typedef __attribute__((ext_vector_type(8)))  __bf16 v8bf;
typedef __attribute__((ext_vector_type(8)))  float  v8f;
typedef int v4i_t __attribute__((vector_size(16)));
typedef __attribute__((address_space(1))) v4i_t* g_v4i_ptr;   // global b128
typedef __attribute__((address_space(3))) v4i_t* l_v4i_ptr;   // LDS b128

// Problem dimensions (fixed by the reference setup)
#define BB   2
#define CC   3
#define FF   51
#define OHH  480
#define OWW  832
#define IHH  (OHH + FF - 1)   // 530
#define IWW  (OWW + FF - 1)   // 882

// Blocking
#define TH     8                    // output rows per block
#define TW     16                   // output cols per block (WMMA N)
#define KREAL  (TW + FF - 1)        // 66 : extended-column GEMM K
#define NKSTEP 3                    // ceil(66/32)
#define APAD   104                  // bf16 row stride (16B aligned, bank-friendly)
#define ARHGT  (FF + TH - 1)        // 58 input rows per channel
#define AROWS  (CC * ARHGT)         // 174
#define MROWS  (CC * FF)            // 153
#define MTILES 10                   // ceil(160/16)
#define HS_STRIDE 17                // hsum f32 row stride (conflict pad)

#define LDS_HS_BYTES  (160 * HS_STRIDE * 4)          // 10880
#define LDS_A_ELEMS   (AROWS * APAD)                 // 18096 bf16 per plane
#define LDS_B_ELEMS   (16 * APAD)                    // 1664 bf16 per plane
#define LDS_VB_BYTES  (FF * TW * 4)                  // 3264 : vert row buffer
#define LDS_BASE      (LDS_HS_BYTES + 2*2*LDS_A_ELEMS + 2*2*LDS_B_ELEMS) // 89920
#define LDS_TOTAL     (LDS_BASE + LDS_VB_BYTES)      // 93184 B

#if __has_builtin(__builtin_amdgcn_global_load_async_to_lds_b128)
#define HAVE_ASYNC_LDS 1
#endif

__global__ __launch_bounds__(128) void sepconv_wmma_kernel(
    const float* __restrict__ inp,
    const float* __restrict__ vert,
    const float* __restrict__ horz,
    float* __restrict__ out)
{
  extern __shared__ char smem[];
  float*  hs   = (float*)smem;                               // [160][17] f32
  bf16_t* Ahi  = (bf16_t*)(smem + LDS_HS_BYTES);             // [174][104]
  bf16_t* Alo  = Ahi + LDS_A_ELEMS;
  bf16_t* Bhi  = Alo + LDS_A_ELEMS;                          // [16][104] banded B^T
  bf16_t* Blo  = Bhi + LDS_B_ELEMS;
  float*  vbuf = (float*)(smem + LDS_BASE);                  // [51][16] f32

  const int tid  = threadIdx.x;
  const int lane = tid & 31;
  const int wave = tid >> 5;
  const int w0 = blockIdx.x * TW;    // 52 tiles
  const int h0 = blockIdx.y * TH;    // 60 tiles
  const int b  = blockIdx.z;         // 2

  // ---- Load input patch -> bf16 hi/lo planes in LDS (once per block) ----
  for (int e = tid; e < AROWS * KREAL; e += 128) {
    int ra = e / KREAL;
    int k  = e - ra * KREAL;
    int c  = ra / ARHGT;
    int r  = ra - c * ARHGT;
    float x = inp[((size_t)(b * CC + c) * IHH + (h0 + r)) * IWW + (w0 + k)];
    bf16_t xh = (bf16_t)x;
    Ahi[ra * APAD + k] = xh;
    Alo[ra * APAD + k] = (bf16_t)(x - (float)xh);
  }
  // Zero K padding (k = 66..103): required so pad-K never injects NaN*0
  for (int e = tid; e < AROWS * (APAD - KREAL); e += 128) {
    int ra = e / (APAD - KREAL);
    int k  = KREAL + (e - ra * (APAD - KREAL));
    Ahi[ra * APAD + k] = (bf16_t)0.0f;
    Alo[ra * APAD + k] = (bf16_t)0.0f;
  }
  // Zero banded-B planes once (band positions are rewritten every row)
  for (int e = tid; e < LDS_B_ELEMS; e += 128) {
    Bhi[e] = (bf16_t)0.0f;
    Blo[e] = (bf16_t)0.0f;
  }

  for (int dh = 0; dh < TH; ++dh) {
    const int h = h0 + dh;
    __syncthreads();  // A ready (first iter); B/hsum/vbuf reusable (later iters)

    // ---- Build banded B^T: Bt[n][n+j] = bf16split(horz[b, j, h, w0+n]) ----
    {
      const int n = tid & 15;
      for (int j = tid >> 4; j < FF; j += 8) {
        const float* gp = &horz[((size_t)(b * FF + j) * OHH + h) * OWW + (w0 + n)];
        float x = *gp;
        if (dh + 1 < TH) __builtin_prefetch(gp + OWW, 0, 1);  // next output row
        bf16_t xh = (bf16_t)x;
        Bhi[n * APAD + (n + j)] = xh;
        Blo[n * APAD + (n + j)] = (bf16_t)(x - (float)xh);
      }
    }

#ifdef HAVE_ASYNC_LDS
    // ---- Async DMA this row's vertical filters into LDS (hidden under WMMA)
    for (int g = tid; g < (FF * TW) / 4; g += 128) {   // 204 groups of 4 floats
      int i  = g >> 2;
      int n4 = (g & 3) << 2;
      const float* gp = &vert[((size_t)(b * FF + i) * OHH + h) * OWW + (w0 + n4)];
      float* lp = vbuf + i * TW + n4;
      __builtin_amdgcn_global_load_async_to_lds_b128(
          (g_v4i_ptr)(unsigned long long)gp,
          (l_v4i_ptr)(unsigned int)(unsigned long long)lp,
          0, 0);
    }
#endif
    __syncthreads();

    // ---- Stage 1 GEMM: hsum[m][n] = sum_k A[m][k] * Bband[k][n] ----
    const int lm   = lane & 15;
    const int half = lane >> 4;

    // Banded-B fragments: identical for every M-tile -> load once per wave
    v16bf bh[NKSTEP], bl[NKSTEP];
    {
      const bf16_t* browh = Bhi + lm * APAD;
      const bf16_t* browl = Blo + lm * APAD;
      #pragma unroll
      for (int s = 0; s < NKSTEP; ++s) {
        const int kb = s * 32 + half * 16;    // B: K 0..15 | 16..31 per half-wave
        *(v8bf*)&bh[s]       = *(const v8bf*)(browh + kb);
        *((v8bf*)&bh[s] + 1) = *(const v8bf*)(browh + kb + 8);
        *(v8bf*)&bl[s]       = *(const v8bf*)(browl + kb);
        *((v8bf*)&bl[s] + 1) = *(const v8bf*)(browl + kb + 8);
      }
    }

    for (int t = wave; t < MTILES; t += 4) {
      v8f acc = {};
      int m = t * 16 + lm;
      if (m > MROWS - 1) m = MROWS - 1;       // clamp padded M rows (unused)
      const int c = m / FF;
      const int i = m - c * FF;
      const bf16_t* arowh = Ahi + (c * ARHGT + i + dh) * APAD;
      const bf16_t* arowl = Alo + (c * ARHGT + i + dh) * APAD;
      #pragma unroll
      for (int s = 0; s < NKSTEP; ++s) {
        const int ka = s * 32 + half * 16;    // A: K 0..15 | 16..31 per half-wave
        v16bf afh, afl;
        *(v8bf*)&afh       = *(const v8bf*)(arowh + ka);
        *((v8bf*)&afh + 1) = *(const v8bf*)(arowh + ka + 8);
        *(v8bf*)&afl       = *(const v8bf*)(arowl + ka);
        *((v8bf*)&afl + 1) = *(const v8bf*)(arowl + ka + 8);
        // fp32-accurate bf16x3: hi*hi + hi*lo + lo*hi (lo*lo ~ 2^-16, dropped)
        acc = __builtin_amdgcn_wmma_f32_16x16x32_bf16(false, afh, false, bh[s], (short)0, acc, false, false);
        acc = __builtin_amdgcn_wmma_f32_16x16x32_bf16(false, afh, false, bl[s], (short)0, acc, false, false);
        acc = __builtin_amdgcn_wmma_f32_16x16x32_bf16(false, afl, false, bh[s], (short)0, acc, false, false);
      }
      // C layout: VGPR r -> M = t*16 + half*8 + r, N = lm
      const int rowbase = t * 16 + half * 8;
      #pragma unroll
      for (int r = 0; r < 8; ++r)
        hs[(rowbase + r) * HS_STRIDE + lm] = acc[r];
    }

#ifdef HAVE_ASYNC_LDS
#if __has_builtin(__builtin_amdgcn_s_wait_asynccnt)
    __builtin_amdgcn_s_wait_asynccnt(0);
#else
    asm volatile("s_wait_asynccnt 0x0" ::: "memory");
#endif
#endif
    __syncthreads();

    // ---- Stage 2: out[c][n] = sum_i vert[i] * hsum[c*F+i][n] ----
    if (tid < CC * TW) {
      const int c = tid >> 4;
      const int n = tid & 15;
      float accv = 0.f;
      for (int i = 0; i < FF; ++i) {
#ifdef HAVE_ASYNC_LDS
        float v = vbuf[i * TW + n];
#else
        const float* vp = &vert[((size_t)(b * FF + i) * OHH + h) * OWW + (w0 + n)];
        if (dh + 1 < TH) __builtin_prefetch(vp + OWW, 0, 1);
        float v = *vp;
#endif
        accv += v * hs[(c * FF + i) * HS_STRIDE + n];
      }
      out[((size_t)(b * CC + c) * OHH + h) * OWW + (w0 + n)] = accv;
    }
  }
}

extern "C" void kernel_launch(void* const* d_in, const int* in_sizes, int n_in,
                              void* d_out, int out_size, void* d_ws, size_t ws_size,
                              hipStream_t stream) {
  (void)in_sizes; (void)n_in; (void)out_size; (void)d_ws; (void)ws_size;
  const float* inp  = (const float*)d_in[0];
  const float* vert = (const float*)d_in[1];
  const float* horz = (const float*)d_in[2];
  float* out = (float*)d_out;
  dim3 grid(OWW / TW, OHH / TH, BB);   // 52 x 60 x 2
  sepconv_wmma_kernel<<<grid, 128, LDS_TOTAL, stream>>>(inp, vert, horz, out);
}
